// selfattention3_43224550867741
// MI455X (gfx1250) — matrix operand order repaired and checked by reading, not compiled, not run on hardware
//
#include <hip/hip_runtime.h>

typedef __attribute__((ext_vector_type(16))) _Float16 v16h;
typedef __attribute__((ext_vector_type(8)))  _Float16 v8h;
typedef __attribute__((ext_vector_type(4)))  _Float16 v4h;
typedef __attribute__((ext_vector_type(8)))  float    v8f;
typedef __attribute__((ext_vector_type(4)))  float    v4f;
typedef __attribute__((ext_vector_type(4)))  unsigned int v4u;
typedef __attribute__((ext_vector_type(8)))  int      v8i;
typedef __attribute__((ext_vector_type(4)))  int      v4i;

#define DIM   512
#define NQKV  1536
#define BATCH 4
#define SEQ   4096
#define ROWS  (BATCH * SEQ)          // 16384 flattened (b,s) rows
#define RSCALE 0.04419417382415922f  // 1/sqrt(512)

static __device__ __forceinline__ v8f wmma_f16(v16h a, v16h b, v8f c) {
  return __builtin_amdgcn_wmma_f32_16x16x32_f16(false, a, false, b, (short)0, c,
                                                false, false);
}

// TDM: 2D tile load global->LDS.  D# packed per CDNA5 ISA ch.8:
//  g0: [1:0]=count=1, [63:32]=lds_addr, [120:64]=global_addr, [127:126]=type 2
//  g1: [17:16]=data_size(1 => 2B), [79:48]=tensor_dim0, [111:80]=tensor_dim1,
//      [127:112]=tile_dim0, [143:128]=tile_dim1, [207:160]=tensor_dim0_stride
static __device__ __forceinline__ void tdm_load_2d(
    unsigned lds_off, unsigned long long gaddr, unsigned dim0, unsigned dim1,
    unsigned tile0, unsigned tile1, unsigned stride0) {
  v4u g0;
  g0[0] = 1u;                                      // count=1, user descriptor
  g0[1] = lds_off;                                 // lds_addr (bytes)
  g0[2] = (unsigned)(gaddr & 0xFFFFFFFFull);       // global_addr[31:0]
  g0[3] = (unsigned)(gaddr >> 32) | (2u << 30);    // addr[56:32] | type=2
  v8i g1;
  g1[0] = (int)(1u << 16);                         // wg_mask=0, data_size=2B
  g1[1] = (int)((dim0 & 0xFFFFu) << 16);           // tensor_dim0[15:0]
  g1[2] = (int)((dim0 >> 16) | ((dim1 & 0xFFFFu) << 16));
  g1[3] = (int)((dim1 >> 16) | ((tile0 & 0xFFFFu) << 16));
  g1[4] = (int)(tile1 & 0xFFFFu);                  // tile_dim1 (tile_dim2=0)
  g1[5] = (int)stride0;                            // tensor_dim0_stride[31:0]
  g1[6] = 0;
  g1[7] = 0;
  v4i z4 = {0, 0, 0, 0};                           // groups 2/3 unused (2D)
  v8i z8 = {0, 0, 0, 0, 0, 0, 0, 0};
  __builtin_amdgcn_tensor_load_to_lds(g0, g1, z4, z4, z8, 0);
}

// ---------------------------------------------------------------------------
// Kernel 1: W [512,1536] fp32 (row-major, x@W) -> Wt [1536,512] f16
// ---------------------------------------------------------------------------
__global__ void wt_convert_kernel(const float* __restrict__ W,
                                  _Float16* __restrict__ Wt) {
  int i = blockIdx.x * 256 + threadIdx.x;
  int n = i >> 9;
  int k = i & 511;
  Wt[i] = (_Float16)W[k * NQKV + n];
}

// ---------------------------------------------------------------------------
// Kernel 2: fused QKV projection.  qkv = x @ W + b (f32 accum via WMMA f16)
//   Stores: Qh[b,s,d] (scaled by 1/sqrt(d)), Kh[b,s,d], Vt[b,d,s]  (all f16)
// ---------------------------------------------------------------------------
__global__ void __launch_bounds__(256) qkv_gemm_kernel(
    const float* __restrict__ x, const _Float16* __restrict__ Wt,
    const float* __restrict__ bias, _Float16* __restrict__ Qh,
    _Float16* __restrict__ Kh, _Float16* __restrict__ Vt) {
  __shared__ _Float16 lx[16 * DIM];
  const int tid = threadIdx.x;
  const int m0  = blockIdx.x * 16;

  const float* xt = x + (size_t)m0 * DIM;
  for (int i = tid; i < 16 * DIM / 4; i += 256) {
    v4f v = *(const v4f*)(xt + 4 * i);
    v4h h;
    h[0] = (_Float16)v[0]; h[1] = (_Float16)v[1];
    h[2] = (_Float16)v[2]; h[3] = (_Float16)v[3];
    *(v4h*)(lx + 4 * i) = h;
  }
  __syncthreads();

  const int lane = tid & 31;
  const int wave = tid >> 5;
  const int half = lane >> 4;
  const int l16  = lane & 15;
  const int n    = blockIdx.y * 128 + wave * 16 + l16;

  v8f c = {};
  const _Float16* wrow = Wt + (size_t)n * DIM;
#pragma unroll
  for (int k0 = 0; k0 < DIM; k0 += 32) {
    const _Float16* ap = lx + l16 * DIM + k0 + (half ? 16 : 0);
    v8h a0 = *(const v8h*)ap;
    v8h a1 = *(const v8h*)(ap + 8);
    v16h a = __builtin_shufflevector(a0, a1, 0, 1, 2, 3, 4, 5, 6, 7, 8, 9, 10,
                                     11, 12, 13, 14, 15);
    v16h b = *(const v16h*)(wrow + k0 + (half ? 16 : 0));
    c = wmma_f16(a, b, c);
  }

  const float bv = bias[n];
  if (n < DIM) {                             // Q (scale folded in)
#pragma unroll
    for (int v = 0; v < 8; ++v) {
      int row = m0 + v + 8 * half;
      Qh[(size_t)row * DIM + n] = (_Float16)((c[v] + bv) * RSCALE);
    }
  } else if (n < 2 * DIM) {                  // K
#pragma unroll
    for (int v = 0; v < 8; ++v) {
      int row = m0 + v + 8 * half;
      Kh[(size_t)row * DIM + (n - DIM)] = (_Float16)(c[v] + bv);
    }
  } else {                                   // V, stored transposed [b,d,s]
    int b   = m0 >> 12;
    int s0  = (m0 & 4095) + 8 * half;
    int dcv = n - 2 * DIM;
    v8h pk;
#pragma unroll
    for (int v = 0; v < 8; ++v) pk[v] = (_Float16)(c[v] + bv);
    *(v8h*)(Vt + ((size_t)b * DIM + dcv) * SEQ + s0) = pk;
  }
}

// ---------------------------------------------------------------------------
// Kernel 3: causal flash attention.
//   block = 256 thr = 8 waves = 4 row-tiles (64 q-rows) x 2 d-slices (256).
//   K/V tiles (32 keys x 512d, 32KB each) TDM-prefetched into LDS, double-
//   buffered; partial scores reduced across the 2 d-slice waves via LDS;
//   streaming softmax in A-fragment domain feeds PV WMMAs directly.
// ---------------------------------------------------------------------------
__global__ void __launch_bounds__(256) attn_kernel(
    const _Float16* __restrict__ Qh, const _Float16* __restrict__ Kh,
    const _Float16* __restrict__ Vt, float* __restrict__ out) {
  extern __shared__ __align__(16) char smem[];
  _Float16* lk = (_Float16*)smem;                      // 2 x [32 keys][512 d]
  _Float16* lv = (_Float16*)(smem + 65536);            // 2 x [512 d][32 keys]
  float*  part = (float*)(smem + 131072);              // 8 x 16x32 partials

  const int tid  = threadIdx.x;
  const int lane = tid & 31;
  const int wave = tid >> 5;
  const int half = lane >> 4;
  const int l16  = lane & 15;
  const int rt   = wave >> 1;                // row-tile 0..3
  const int dsl  = wave & 1;                 // d-slice 0..1
  const int d0   = dsl << 8;                 // 0 or 256

  // pair-swizzle query groups so co-resident blocks have complementary work
  const int b   = blockIdx.x >> 6;
  const int t6  = blockIdx.x & 63;
  const int g   = (t6 & 1) ? (63 - (t6 >> 1)) : (t6 >> 1);
  const int qb0 = g << 6;                    // 64 q-rows per block
  const int qi0 = qb0 + rt * 16;             // this wave's 16 q-rows

  const unsigned lkoff = (unsigned)(size_t)lk;
  const unsigned lvoff = (unsigned)(size_t)lv;
  const unsigned long long kgbase =
      (unsigned long long)(size_t)(Kh + (size_t)b * SEQ * DIM);
  const unsigned long long vgbase =
      (unsigned long long)(size_t)(Vt + (size_t)b * DIM * SEQ);

  // Q fragments for this wave's 256-wide d-slice (A layout, row = l16)
  const _Float16* qrow = Qh + ((size_t)(b * SEQ + qi0 + l16)) * DIM + d0;
  v16h qf[8];
#pragma unroll
  for (int kk = 0; kk < 8; ++kk) {
    const _Float16* ap = qrow + kk * 32 + (half ? 16 : 0);
    v8h a0 = *(const v8h*)ap;
    v8h a1 = *(const v8h*)(ap + 8);
    qf[kk] = __builtin_shufflevector(a0, a1, 0, 1, 2, 3, 4, 5, 6, 7, 8, 9, 10,
                                     11, 12, 13, 14, 15);
  }

  v8f oa[16];
#pragma unroll
  for (int t = 0; t < 16; ++t) oa[t] = (v8f){};
  float m_r = -3.0e38f, l_r = 0.0f;

  float* myp = part + (size_t)(rt * 2 + dsl) * 512;
  const int nj = qb0 + 64;                   // exclusive causal key bound
  const int cA = half ? 8 : 0;

  // prologue: prefetch first K/V tile into buffer 0 (TDM, wave 0 only)
  if (wave == 0) {
    tdm_load_2d(lkoff, kgbase, DIM, SEQ, DIM, 32, DIM);
    tdm_load_2d(lvoff, vgbase, SEQ, DIM, 32, DIM, SEQ);
  }

  int cur = 0;
  for (int j0 = 0; j0 < nj; j0 += 32, cur ^= 1) {
    // ---- prefetch next tile into the other buffer, wait for current one
    if (wave == 0) {
      if (j0 + 32 < nj) {
        unsigned nb = (unsigned)(cur ^ 1) * 32768u;
        tdm_load_2d(lkoff + nb, kgbase + (size_t)(j0 + 32) * DIM * 2, DIM, SEQ,
                    DIM, 32, DIM);
        tdm_load_2d(lvoff + nb, vgbase + (size_t)(j0 + 32) * 2, SEQ, DIM, 32,
                    DIM, SEQ);
        __builtin_amdgcn_s_wait_tensorcnt(2);  // in-order: current tile done
      } else {
        __builtin_amdgcn_s_wait_tensorcnt(0);
      }
    }
    __syncthreads();                         // publish buf[cur]

    // ---- partial S for 32 keys from LDS K tile (two 16x16 C tiles)
    v8f s0 = {}, s1 = {};
    const _Float16* kb = lk + (size_t)cur * (32 * 512) + (half ? 16 : 0);
#pragma unroll
    for (int kk = 0; kk < 8; ++kk) {
      int dofs = d0 + kk * 32;
      v16h b0 = *(const v16h*)(kb + (size_t)l16 * 512 + dofs);
      v16h b1 = *(const v16h*)(kb + (size_t)(l16 + 16) * 512 + dofs);
      s0 = wmma_f16(qf[kk], b0, s0);
      s1 = wmma_f16(qf[kk], b1, s1);
    }
    // write partials (C layout: row = v + 8*half, col = l16 / l16+16)
#pragma unroll
    for (int v = 0; v < 8; ++v) {
      int row = v + 8 * half;
      myp[row * 32 + l16]      = s0[v];
      myp[row * 32 + l16 + 16] = s1[v];
    }
    __syncthreads();

    // ---- sum the 2 d-slice partials, read in A-fragment order
    float acc[16];
#pragma unroll
    for (int i = 0; i < 16; ++i) acc[i] = 0.0f;
#pragma unroll
    for (int w = 0; w < 2; ++w) {
      const float* rp = part + (size_t)(rt * 2 + w) * 512 + l16 * 32 + cA;
      v4f u0 = *(const v4f*)(rp + 0);
      v4f u1 = *(const v4f*)(rp + 4);
      v4f u2 = *(const v4f*)(rp + 16);
      v4f u3 = *(const v4f*)(rp + 20);
#pragma unroll
      for (int i = 0; i < 4; ++i) {
        acc[i]      += u0[i];
        acc[4 + i]  += u1[i];
        acc[8 + i]  += u2[i];
        acc[12 + i] += u3[i];
      }
    }

    // ---- causal mask
    const int qrg = qi0 + l16;
#pragma unroll
    for (int i = 0; i < 8; ++i) {
      acc[i]     = (j0 + cA + i      > qrg) ? -3.0e38f : acc[i];
      acc[8 + i] = (j0 + cA + 16 + i > qrg) ? -3.0e38f : acc[8 + i];
    }

    // ---- streaming softmax (row = this lane + lane^16 partner)
    float mb = acc[0];
#pragma unroll
    for (int i = 1; i < 16; ++i) mb = fmaxf(mb, acc[i]);
    mb = fmaxf(mb, __shfl_xor(mb, 16, 32));
    const float mn = fmaxf(m_r, mb);
    float p[16], rs = 0.0f;
#pragma unroll
    for (int i = 0; i < 16; ++i) { p[i] = __expf(acc[i] - mn); rs += p[i]; }
    rs += __shfl_xor(rs, 16, 32);
    const float alpha = __expf(m_r - mn);
    l_r = l_r * alpha + rs;
    m_r = mn;

    v16h pf;                                 // P as f16 A-fragment
#pragma unroll
    for (int i = 0; i < 16; ++i) pf[i] = (_Float16)p[i];

    // ---- rescale O (broadcast row-alpha lane-domain -> C-tile domain)
#pragma unroll
    for (int v = 0; v < 8; ++v) {
      float sc = __shfl(alpha, v + (half ? 8 : 0), 32);
#pragma unroll
      for (int t = 0; t < 16; ++t) oa[t][v] *= sc;
    }

    // ---- O += P @ V from LDS V tile [d][32 keys]
    const _Float16* vb = lv + (size_t)cur * (512 * 32) + (half ? 8 : 0);
#pragma unroll
    for (int t = 0; t < 16; ++t) {
      int dcol = d0 + t * 16 + l16;
      v8h v0 = *(const v8h*)(vb + (size_t)dcol * 32);
      v8h v1 = *(const v8h*)(vb + (size_t)dcol * 32 + 16);
      v16h vv = __builtin_shufflevector(v0, v1, 0, 1, 2, 3, 4, 5, 6, 7, 8, 9, 10,
                                        11, 12, 13, 14, 15);
      oa[t] = wmma_f16(pf, vv, oa[t]);
    }
    __syncthreads();                         // buf[cur] + partials reusable
  }

  // ---- epilogue: out = O / l   (fp32)
  const float rl = 1.0f / l_r;
  float* ob = out + ((size_t)(b * SEQ + qi0)) * DIM + d0 + l16;
#pragma unroll
  for (int v = 0; v < 8; ++v) {
    float sc = __shfl(rl, v + (half ? 8 : 0), 32);
    int M = v + 8 * half;
#pragma unroll
    for (int t = 0; t < 16; ++t) ob[(size_t)M * DIM + t * 16] = oa[t][v] * sc;
  }
}

// ---------------------------------------------------------------------------
extern "C" void kernel_launch(void* const* d_in, const int* in_sizes, int n_in,
                              void* d_out, int out_size, void* d_ws,
                              size_t ws_size, hipStream_t stream) {
  (void)in_sizes; (void)n_in; (void)out_size; (void)ws_size;
  const float* x    = (const float*)d_in[0];   // [4,4096,512]
  const float* W    = (const float*)d_in[1];   // [512,1536]
  const float* bqkv = (const float*)d_in[2];   // [1536]
  float* out        = (float*)d_out;           // [4,4096,512]

  char* ws = (char*)d_ws;
  const size_t WT_OFF = 0;
  const size_t QH_OFF = WT_OFF + (size_t)NQKV * DIM * sizeof(_Float16);
  const size_t KH_OFF = QH_OFF + (size_t)ROWS * DIM * sizeof(_Float16);
  const size_t VT_OFF = KH_OFF + (size_t)ROWS * DIM * sizeof(_Float16);
  _Float16* Wt = (_Float16*)(ws + WT_OFF);
  _Float16* Qh = (_Float16*)(ws + QH_OFF);
  _Float16* Kh = (_Float16*)(ws + KH_OFF);
  _Float16* Vt = (_Float16*)(ws + VT_OFF);

  wt_convert_kernel<<<(NQKV * DIM) / 256, 256, 0, stream>>>(W, Wt);
  dim3 ggrid(ROWS / 16, NQKV / 128);
  qkv_gemm_kernel<<<ggrid, 256, 0, stream>>>(x, Wt, bqkv, Qh, Kh, Vt);
  // 64 q-rows per block, 144KB dynamic LDS (2x32KB K + 2x32KB V + 16KB parts)
  attn_kernel<<<BATCH * (SEQ / 64), 256, 147456, stream>>>(Qh, Kh, Vt, out);
}
